// VerticalModel_GIN_GT_CombinedEmbeds_71665824301900
// MI455X (gfx1250) — compile-verified, matrix operand order checked by reading, not run on hardware
//
#include <hip/hip_runtime.h>
#include <hip/hip_bf16.h>
#include <math.h>

// ---------------- problem constants (match reference) ----------------
#define F_IN    128
#define HID     64
#define HEADS   4
#define EDIM    16
#define GG      128
#define TT      8
#define HC      256   // HID*HEADS

typedef __attribute__((ext_vector_type(16))) __bf16 v16bf;
typedef __attribute__((ext_vector_type(8)))  float  v8f;

// ---------------- small utility kernels ----------------
__global__ void fill_kernel(float* p, float v, long long n) {
  long long i = (long long)blockIdx.x * blockDim.x + threadIdx.x;
  if (i < n) p[i] = v;
}

__device__ inline void atomicMaxF(float* addr, float val) {
  if (val >= 0.f) atomicMax((int*)addr, __float_as_int(val));
  else            atomicMin((unsigned int*)addr, __float_as_uint(val));
}

// ---------------- GIN: scatter-add of h[src] into agg[dst] ----------------
__global__ void scatter_add_kernel(const float* __restrict__ h,
                                   const int* __restrict__ src,
                                   const int* __restrict__ dst,
                                   float* __restrict__ agg,
                                   int F, long long total) {
  long long i = (long long)blockIdx.x * blockDim.x + threadIdx.x;
  if (i >= total) return;
  int e = (int)(i / F);
  int f = (int)(i % F);
  atomicAdd(&agg[(long long)dst[e] * F + f], h[(long long)src[e] * F + f]);
}

// ---------------- weight pre-pack into WMMA B-fragment layout (bf16) -------
// Fragment order: [ks][nt][lane][elem], elem e of lane L maps to
// K = ks*32 + (e/8)*16 + (L/16)*8 + e%8,  N = nt*16 + L%16.
__global__ void pack_w_kernel(const float* __restrict__ W, __bf16* __restrict__ Wpk,
                              int K, int Nout) {
  int i = blockIdx.x * blockDim.x + threadIdx.x;
  if (i >= K * Nout) return;
  const int e    = i & 15;
  const int lane = (i >> 4) & 31;
  const int NT   = Nout >> 4;
  const int nt   = (i >> 9) % NT;
  const int ks   = i / (NT << 9);
  const int kk   = (ks << 5) + (((e >> 3) << 4) | ((lane >> 4) << 3) | (e & 7));
  const int col  = (nt << 4) + (lane & 15);
  Wpk[i] = (__bf16)W[(long long)kk * Nout + col];
}

// ---------------- WMMA bf16 GEMM: Out = act((A [+A2]) @ W + bias) ----------
// A: M x K fp32 (M multiple of 16, K multiple of 32), Wpk: packed bf16.
// block = 128 threads (4 waves); wave -> N tile, block.x -> 4 M subtiles.
__device__ inline void cvt8(v16bf& a, int base, float4 x, float4 y) {
  a[base + 0] = (__bf16)x.x; a[base + 1] = (__bf16)x.y;
  a[base + 2] = (__bf16)x.z; a[base + 3] = (__bf16)x.w;
  a[base + 4] = (__bf16)y.x; a[base + 5] = (__bf16)y.y;
  a[base + 6] = (__bf16)y.z; a[base + 7] = (__bf16)y.w;
}

template <bool HAS_A2>
__global__ void gemm_wmma_kernel(const float* __restrict__ A,
                                 const float* __restrict__ A2,
                                 const __bf16* __restrict__ Wpk,
                                 const float* __restrict__ bias,
                                 float* __restrict__ Out,
                                 int M, int K, int Nout, int relu) {
  const int lane  = threadIdx.x & 31;
  const int wave  = threadIdx.x >> 5;
  const int NT    = Nout >> 4;
  const int nt    = blockIdx.y * 4 + wave;
  if (nt >= NT) return;                       // wave-uniform exit
  const int mbase = blockIdx.x * 64;
  const int col   = (nt << 4) + (lane & 15);
  const int khi   = (lane >> 4) << 3;
  const int KS    = K >> 5;

  v8f acc0 = {}, acc1 = {}, acc2 = {}, acc3 = {};
  for (int ks = 0; ks < KS; ++ks) {
    const v16bf b = *(const v16bf*)(Wpk + (((long long)(ks * NT + nt)) << 9) + (lane << 4));
#pragma unroll
    for (int s = 0; s < 4; ++s) {
      if (mbase + s * 16 < M) {               // scalar (block-uniform) guard
        const int row = mbase + s * 16 + (lane & 15);
        const float4* pa = (const float4*)(A + (long long)row * K + (ks << 5) + khi);
        float4 x0 = pa[0], x1 = pa[1];        // K offsets +0..7
        float4 x2 = pa[4], x3 = pa[5];        // K offsets +16..23
        if (HAS_A2) {
          const float4* p2 = (const float4*)(A2 + (long long)row * K + (ks << 5) + khi);
          float4 y0 = p2[0], y1 = p2[1], y2 = p2[4], y3 = p2[5];
          x0.x += y0.x; x0.y += y0.y; x0.z += y0.z; x0.w += y0.w;
          x1.x += y1.x; x1.y += y1.y; x1.z += y1.z; x1.w += y1.w;
          x2.x += y2.x; x2.y += y2.y; x2.z += y2.z; x2.w += y2.w;
          x3.x += y3.x; x3.y += y3.y; x3.z += y3.z; x3.w += y3.w;
        }
        v16bf a;
        cvt8(a, 0, x0, x1);
        cvt8(a, 8, x2, x3);
        v8f* accp = (s == 0) ? &acc0 : (s == 1) ? &acc1 : (s == 2) ? &acc2 : &acc3;
        *accp = __builtin_amdgcn_wmma_f32_16x16x32_bf16(
            false, a, false, b, (short)0, *accp, false, false);
      }
    }
  }

  const float bv = bias ? bias[col] : 0.f;
#pragma unroll
  for (int s = 0; s < 4; ++s) {
    if (mbase + s * 16 < M) {
      const v8f accv = (s == 0) ? acc0 : (s == 1) ? acc1 : (s == 2) ? acc2 : acc3;
#pragma unroll
      for (int r = 0; r < 8; ++r) {
        const int orow = mbase + s * 16 + r + khi;
        float v = accv[r] + bv;
        if (relu) v = fmaxf(v, 0.f);
        Out[(long long)orow * Nout + col] = v;
      }
    }
  }
}

// ---------------- GT edge kernels ----------------
__global__ void gt_edge_logits_kernel(const float* __restrict__ q,
                                      const float* __restrict__ k,
                                      const float* __restrict__ We,
                                      const float* __restrict__ be,
                                      const float* __restrict__ ea,
                                      const int* __restrict__ src,
                                      const int* __restrict__ dst,
                                      float* __restrict__ logits,
                                      float* __restrict__ seg_m) {
  __shared__ float sea[EDIM];
  __shared__ float red[HC];
  const int e = blockIdx.x;
  const int j = threadIdx.x;                   // 0..255
  if (j < EDIM) sea[j] = ea[(long long)e * EDIM + j];
  __syncthreads();
  const int s = src[e], d = dst[e];
  float ev = be[j];
#pragma unroll
  for (int t = 0; t < EDIM; ++t) ev += sea[t] * We[t * HC + j];
  const float ke = k[(long long)s * HC + j] + ev;
  red[j] = q[(long long)d * HC + j] * ke;
  __syncthreads();
  if (j < HEADS) {
    float sum = 0.f;
    for (int t = 0; t < HID; ++t) sum += red[j * HID + t];
    const float logit = sum * 0.125f;          // 1/sqrt(HID)
    logits[(long long)e * HEADS + j] = logit;
    atomicMaxF(&seg_m[d * HEADS + j], logit);
  }
}

__global__ void seg_exp_sum_kernel(float* __restrict__ alpha,   // in: logits, out: exp
                                   const int* __restrict__ dst,
                                   const float* __restrict__ seg_m,
                                   float* __restrict__ seg_s,
                                   long long total) {
  long long i = (long long)blockIdx.x * blockDim.x + threadIdx.x;
  if (i >= total) return;
  const int e = (int)(i >> 2), h = (int)(i & 3);
  const int d = dst[e];
  const float ex = __expf(alpha[i] - seg_m[d * HEADS + h]);
  alpha[i] = ex;
  atomicAdd(&seg_s[d * HEADS + h], ex);
}

__global__ void gt_edge_message_kernel(const float* __restrict__ v,
                                       const float* __restrict__ We,
                                       const float* __restrict__ be,
                                       const float* __restrict__ ea,
                                       const int* __restrict__ src,
                                       const int* __restrict__ dst,
                                       const float* __restrict__ alpha,
                                       const float* __restrict__ seg_s,
                                       float* __restrict__ out) {
  __shared__ float sea[EDIM];
  const int e = blockIdx.x;
  const int j = threadIdx.x;
  if (j < EDIM) sea[j] = ea[(long long)e * EDIM + j];
  __syncthreads();
  const int s = src[e], d = dst[e];
  float ev = be[j];
#pragma unroll
  for (int t = 0; t < EDIM; ++t) ev += sea[t] * We[t * HC + j];
  const int h = j >> 6;
  const float a = alpha[(long long)e * HEADS + h] / (seg_s[d * HEADS + h] + 1e-16f);
  atomicAdd(&out[(long long)d * HC + j], (v[(long long)s * HC + j] + ev) * a);
}

__global__ void gt_beta_combine_kernel(const float* __restrict__ out,
                                       const float* __restrict__ xr,
                                       const float* __restrict__ Wb,
                                       float* __restrict__ g) {
  __shared__ float red[HC];
  const int n = blockIdx.x, j = threadIdx.x;
  const float o = out[(long long)n * HC + j];
  const float x = xr[(long long)n * HC + j];
  red[j] = o * Wb[j] + x * Wb[HC + j] + (o - x) * Wb[2 * HC + j];
  __syncthreads();
  for (int st = 128; st > 0; st >>= 1) {
    if (j < st) red[j] += red[j + st];
    __syncthreads();
  }
  const float beta = 1.f / (1.f + __expf(-red[0]));
  g[(long long)n * HC + j] = beta * x + (1.f - beta) * o;
}

// ---------------- pooling / readout ----------------
__global__ void pool_add_kernel(const float* __restrict__ h,
                                const int* __restrict__ batch,
                                float* __restrict__ pooled,
                                int F, long long total) {
  long long i = (long long)blockIdx.x * blockDim.x + threadIdx.x;
  if (i >= total) return;
  const int n = (int)(i / F), f = (int)(i % F);
  atomicAdd(&pooled[batch[n] * F + f], h[i]);
}

__global__ void count_kernel(const int* __restrict__ batch, float* __restrict__ cnt, int n) {
  int i = blockIdx.x * blockDim.x + threadIdx.x;
  if (i < n) atomicAdd(&cnt[batch[i]], 1.f);
}

__global__ void finalize_z_kernel(const float* __restrict__ pg,
                                  const float* __restrict__ pt,
                                  const float* __restrict__ cnt,
                                  float* __restrict__ z) {
  int i = blockIdx.x * blockDim.x + threadIdx.x;
  if (i >= GG * (HID + HC)) return;
  const int g = i / (HID + HC), c = i % (HID + HC);
  if (c < HID) z[i] = pg[g * HID + c];
  else         z[i] = pt[g * HC + (c - HID)] / fmaxf(cnt[g], 1.f);
}

__global__ void ro2_kernel(const float* __restrict__ hid,
                           const float* __restrict__ W2,
                           const float* __restrict__ b2,
                           float* __restrict__ out) {
  int i = blockIdx.x * blockDim.x + threadIdx.x;
  if (i >= GG * TT) return;
  const int g = i / TT, t = i % TT;
  float s = b2[t];
  for (int k = 0; k < HID; ++k) s += hid[g * HID + k] * W2[k * TT + t];
  out[i] = s;
}

// ---------------- host-side launch helpers ----------------
static inline void launch_fill(float* p, float v, long long n, hipStream_t s) {
  fill_kernel<<<(int)((n + 255) / 256), 256, 0, s>>>(p, v, n);
}
static inline void launch_gemm(const float* A, const float* A2, const float* W,
                               __bf16* wpk, const float* bias, float* Out,
                               int M, int K, int Nout, int relu, hipStream_t s) {
  pack_w_kernel<<<(K * Nout + 255) / 256, 256, 0, s>>>(W, wpk, K, Nout);
  dim3 grid((M + 63) / 64, (Nout + 63) / 64);
  if (A2) gemm_wmma_kernel<true ><<<grid, 128, 0, s>>>(A, A2, wpk, bias, Out, M, K, Nout, relu);
  else    gemm_wmma_kernel<false><<<grid, 128, 0, s>>>(A, A2, wpk, bias, Out, M, K, Nout, relu);
}

extern "C" void kernel_launch(void* const* d_in, const int* in_sizes, int n_in,
                              void* d_out, int out_size, void* d_ws, size_t ws_size,
                              hipStream_t stream) {
  const float* x    = (const float*)d_in[0];           // N x F_IN
  const float* ea   = (const float*)d_in[1];           // E x EDIM
  const int*   eidx = (const int*)d_in[2];             // 2 x E
  const int*   batch= (const int*)d_in[3];             // N
  const int E = in_sizes[1] / EDIM;
  const int N = in_sizes[0] / F_IN;
  const int* src = eidx;
  const int* dst = eidx + E;

  // ---- params: JAX pytree flatten order (dict keys sorted; None dropped) ----
  int pi = 4;
  const float* ginW1[3]; const float* ginB1[3];
  const float* ginW2[3]; const float* ginB2[3];
  for (int l = 0; l < 3; ++l) {
    ginW1[l] = (const float*)d_in[pi++]; ginB1[l] = (const float*)d_in[pi++];
    ginW2[l] = (const float*)d_in[pi++]; ginB2[l] = (const float*)d_in[pi++];
  }
  struct GTP { const float *betaW,*eW,*eB,*kW,*kB,*qW,*qB,*skipW,*skipB,*vW,*vB; } gt[2];
  for (int l = 0; l < 2; ++l) {  // keys sorted: beta, e, k, q, skip, v (W before b)
    gt[l].betaW = (const float*)d_in[pi++];
    gt[l].eW    = (const float*)d_in[pi++]; gt[l].eB    = (const float*)d_in[pi++];
    gt[l].kW    = (const float*)d_in[pi++]; gt[l].kB    = (const float*)d_in[pi++];
    gt[l].qW    = (const float*)d_in[pi++]; gt[l].qB    = (const float*)d_in[pi++];
    gt[l].skipW = (const float*)d_in[pi++]; gt[l].skipB = (const float*)d_in[pi++];
    gt[l].vW    = (const float*)d_in[pi++]; gt[l].vB    = (const float*)d_in[pi++];
  }
  const float* ro1W = (const float*)d_in[pi++]; const float* ro1B = (const float*)d_in[pi++];
  const float* ro2W = (const float*)d_in[pi++]; const float* ro2B = (const float*)d_in[pi++];

  // ---- workspace carve-up ----
  float* ws = (float*)d_ws;
  float* gin_agg = ws;                   ws += (long long)N * F_IN;
  float* gin_t   = ws;                   ws += (long long)N * HID;
  float* gin_h   = ws;                   ws += (long long)N * HID;
  float* gt_g    = ws;                   ws += (long long)N * HC;
  float* qb      = ws;                   ws += (long long)N * HC;
  float* kb      = ws;                   ws += (long long)N * HC;
  float* vb      = ws;                   ws += (long long)N * HC;
  float* skb     = ws;                   ws += (long long)N * HC;
  float* attout  = ws;                   ws += (long long)N * HC;
  float* alpha   = ws;                   ws += (long long)E * HEADS;
  float* seg_m   = ws;                   ws += (long long)N * HEADS;
  float* seg_s   = ws;                   ws += (long long)N * HEADS;
  float* pool_g  = ws;                   ws += (long long)GG * HID;
  float* pool_t  = ws;                   ws += (long long)GG * HC;
  float* cnt     = ws;                   ws += GG;
  float* zcat    = ws;                   ws += (long long)GG * (HID + HC);
  float* hid     = ws;                   ws += (long long)GG * HID;
  __bf16* wpk    = (__bf16*)ws;          ws += (HC * HC) / 2 + 64;   // packed weight buffer

  // ================= GIN layers =================
  launch_fill(gin_agg, 0.f, (long long)N * F_IN, stream);
  {
    long long tot = (long long)E * F_IN;
    scatter_add_kernel<<<(int)((tot + 255) / 256), 256, 0, stream>>>(x, src, dst, gin_agg, F_IN, tot);
  }
  launch_gemm(gin_agg, x, ginW1[0], wpk, ginB1[0], gin_t, N, F_IN, HID, 1, stream);
  launch_gemm(gin_t, nullptr, ginW2[0], wpk, ginB2[0], gin_h, N, HID, HID, 0, stream);
  for (int l = 1; l < 3; ++l) {
    launch_fill(gin_agg, 0.f, (long long)N * HID, stream);
    long long tot = (long long)E * HID;
    scatter_add_kernel<<<(int)((tot + 255) / 256), 256, 0, stream>>>(gin_h, src, dst, gin_agg, HID, tot);
    launch_gemm(gin_agg, gin_h, ginW1[l], wpk, ginB1[l], gin_t, N, HID, HID, 1, stream);
    launch_gemm(gin_t, nullptr, ginW2[l], wpk, ginB2[l], gin_h, N, HID, HID, 0, stream);
  }
  launch_fill(pool_g, 0.f, (long long)GG * HID, stream);
  {
    long long tot = (long long)N * HID;
    pool_add_kernel<<<(int)((tot + 255) / 256), 256, 0, stream>>>(gin_h, batch, pool_g, HID, tot);
  }

  // ================= GT layers =================
  for (int l = 0; l < 2; ++l) {
    const float* g_in = (l == 0) ? gin_h : gt_g;
    const int fi = (l == 0) ? HID : HC;
    launch_gemm(g_in, nullptr, gt[l].qW,    wpk, gt[l].qB,    qb,  N, fi, HC, 0, stream);
    launch_gemm(g_in, nullptr, gt[l].kW,    wpk, gt[l].kB,    kb,  N, fi, HC, 0, stream);
    launch_gemm(g_in, nullptr, gt[l].vW,    wpk, gt[l].vB,    vb,  N, fi, HC, 0, stream);
    launch_gemm(g_in, nullptr, gt[l].skipW, wpk, gt[l].skipB, skb, N, fi, HC, 0, stream);

    launch_fill(seg_m, -INFINITY, (long long)N * HEADS, stream);
    launch_fill(seg_s, 0.f,       (long long)N * HEADS, stream);
    launch_fill(attout, 0.f,      (long long)N * HC,    stream);

    gt_edge_logits_kernel<<<E, HC, 0, stream>>>(qb, kb, gt[l].eW, gt[l].eB, ea, src, dst, alpha, seg_m);
    {
      long long tot = (long long)E * HEADS;
      seg_exp_sum_kernel<<<(int)((tot + 255) / 256), 256, 0, stream>>>(alpha, dst, seg_m, seg_s, tot);
    }
    gt_edge_message_kernel<<<E, HC, 0, stream>>>(vb, gt[l].eW, gt[l].eB, ea, src, dst, alpha, seg_s, attout);
    gt_beta_combine_kernel<<<N, HC, 0, stream>>>(attout, skb, gt[l].betaW, gt_g);
  }

  // ================= pooling + readout =================
  launch_fill(pool_t, 0.f, (long long)GG * HC, stream);
  launch_fill(cnt,    0.f, GG, stream);
  {
    long long tot = (long long)N * HC;
    pool_add_kernel<<<(int)((tot + 255) / 256), 256, 0, stream>>>(gt_g, batch, pool_t, HC, tot);
  }
  count_kernel<<<(N + 255) / 256, 256, 0, stream>>>(batch, cnt, N);
  finalize_z_kernel<<<(GG * (HID + HC) + 255) / 256, 256, 0, stream>>>(pool_g, pool_t, cnt, zcat);

  launch_gemm(zcat, nullptr, ro1W, wpk, ro1B, hid, GG, HID + HC, HID, 1, stream);
  ro2_kernel<<<(GG * TT + 255) / 256, 256, 0, stream>>>(hid, ro2W, ro2B, (float*)d_out);
}
